// BiAttention_64630667870627
// MI455X (gfx1250) — matrix operand order repaired and checked by reading, not compiled, not run on hardware
//
#include <hip/hip_runtime.h>
#include <hip/hip_bf16.h>
#include <math.h>

// ---------------------------------------------------------------------------
// CDNA5 (gfx1250) implementation: all GEMMs via v_wmma_f32_16x16x32_f16.
// fp32 inputs converted to f16 at load; fp32 accumulate; fp32 glue kernels.
// Wave tile = 16x64 (4 WMMA accumulators share one A fragment per K-step).
// B tile (32K x 64N) is always staged in LDS as f16, K-transposed, so every
// WMMA B-fragment is one contiguous 32-byte LDS read per lane.
// ---------------------------------------------------------------------------

typedef __attribute__((ext_vector_type(16))) _Float16 v16h;
typedef __attribute__((ext_vector_type(2)))  _Float16 v2h;
typedef __attribute__((ext_vector_type(8)))  float    v8f;
typedef unsigned char mask_t;   // jax bool_ arrays: 1 byte per element

// --- A fragment loader (contiguous k order, same as the B fragment) ----------
// A 16x32 f16 fragment from row-major fp32 [.. x lda], tile row origin Arow.
// lanes 0-15: M=lane, K = 0..15; lanes 16-31: M=lane-16, K = 16..31
__device__ __forceinline__ v16h load_a_frag(const float* __restrict__ Arow,
                                            int lda, int kbase, int lane) {
  int m = lane & 15;
  int koff = (lane >> 4) << 4;               // 0 or 16
  const float* p = Arow + (long)m * lda + kbase + koff;
  v16h a;
#pragma unroll
  for (int v = 0; v < 8; ++v) {
    int k0 = 2 * v;
    float2 f = *(const float2*)(p + k0);
    a[2 * v]     = (_Float16)f.x;
    a[2 * v + 1] = (_Float16)f.y;
  }
  return a;
}

// --- generic batched WMMA GEMM:  C = relu?(alpha * A*B(^T) + bias) ----------
// grid: x = N/64, y = M/64, z = batch.  block = 128 (4 waves along M).
// Each wave owns a 16x64 C tile: 4 accumulators, one A fragment per K-step.
template <int TRANSB>
__global__ void __launch_bounds__(128)
wmma_gemm(const float* __restrict__ A, const float* __restrict__ B,
          const float* __restrict__ bias, float* __restrict__ C,
          int K, long strideA, long strideB, long strideC,
          int lda, int ldb, int ldc, float alpha, int relu) {
  // ldsB[n][k] = B[kb+k][n0+n] (f16): per-column K-contiguous -> one 32B read
  // per lane per fragment.
  __shared__ alignas(32) _Float16 ldsB[64][32];   // 4 KB

  int lane = threadIdx.x & 31;
  int wave = threadIdx.x >> 5;
  int m0 = blockIdx.y * 64 + wave * 16;
  int n0 = blockIdx.x * 64;

  const float* Arow = A + (long)blockIdx.z * strideA + (long)m0 * lda;
  const float* Bb   = B + (long)blockIdx.z * strideB;

  int nloc = lane & 15;
  int kh   = (lane >> 4) << 4;        // 0 or 16

  v8f acc[4] = {};
  for (int kb = 0; kb < K; kb += 32) {
    // prefetch next K-slice of this wave's A rows (global_prefetch_b8)
    if (kb + 32 < K)
      __builtin_prefetch(Arow + (long)(lane & 15) * lda + kb + 32, 0, 3);

    // A fragment: issue early so global latency overlaps LDS staging.
    v16h a = load_a_frag(Arow, lda, kb, lane);

    if (kb) __syncthreads();          // previous-step LDS reads complete

    if (TRANSB) {
      // src is [N x ldb(=ldk)]: rows are K-contiguous. Thread -> 64B row seg.
      int n    = threadIdx.x >> 1;          // 0..63
      int half = (threadIdx.x & 1) << 4;    // 0 or 16
      const float* p = Bb + (long)(n0 + n) * ldb + kb + half;
      float4 f0 = *(const float4*)(p);
      float4 f1 = *(const float4*)(p + 4);
      float4 f2 = *(const float4*)(p + 8);
      float4 f3 = *(const float4*)(p + 12);
      v16h hv;
      hv[0]  = (_Float16)f0.x; hv[1]  = (_Float16)f0.y;
      hv[2]  = (_Float16)f0.z; hv[3]  = (_Float16)f0.w;
      hv[4]  = (_Float16)f1.x; hv[5]  = (_Float16)f1.y;
      hv[6]  = (_Float16)f1.z; hv[7]  = (_Float16)f1.w;
      hv[8]  = (_Float16)f2.x; hv[9]  = (_Float16)f2.y;
      hv[10] = (_Float16)f2.z; hv[11] = (_Float16)f2.w;
      hv[12] = (_Float16)f3.x; hv[13] = (_Float16)f3.y;
      hv[14] = (_Float16)f3.z; hv[15] = (_Float16)f3.w;
      *(v16h*)&ldsB[n][half] = hv;
    } else {
      // src is [K x ldb]: coalesced float4 along N; thread -> (K pair, 8 cols)
      int k0   = (threadIdx.x >> 3) * 2;    // 0,2,..,30
      int cgrp = (threadIdx.x & 7) * 8;     // 0,8,..,56
      const float* r0 = Bb + (long)(kb + k0) * ldb + n0 + cgrp;
      const float* r1 = r0 + ldb;
      float4 a0 = *(const float4*)(r0);
      float4 a1 = *(const float4*)(r0 + 4);
      float4 b0 = *(const float4*)(r1);
      float4 b1 = *(const float4*)(r1 + 4);
      float r0f[8] = {a0.x, a0.y, a0.z, a0.w, a1.x, a1.y, a1.z, a1.w};
      float r1f[8] = {b0.x, b0.y, b0.z, b0.w, b1.x, b1.y, b1.z, b1.w};
#pragma unroll
      for (int i = 0; i < 8; ++i) {
        v2h h;
        h[0] = (_Float16)r0f[i];      // B[kb+k0  ][n0+cgrp+i]
        h[1] = (_Float16)r1f[i];      // B[kb+k0+1][n0+cgrp+i]
        *(v2h*)&ldsB[cgrp + i][k0] = h;
      }
    }
    __syncthreads();

    v16h bf[4];
#pragma unroll
    for (int j = 0; j < 4; ++j)
      bf[j] = *(const v16h*)&ldsB[16 * j + nloc][kh];
#pragma unroll
    for (int j = 0; j < 4; ++j)
      acc[j] = __builtin_amdgcn_wmma_f32_16x16x32_f16(false, a, false, bf[j],
                                                      (short)0, acc[j],
                                                      false, false);
  }

  // C/D layout: VGPR r -> row r (lanes 0-15) / row 8+r (lanes 16-31), col=lane&15
  int col = lane & 15;
  int rbase = (lane >> 4) << 3;
  float* Cbase = C + (long)blockIdx.z * strideC + (long)(m0 + rbase) * ldc + col;
#pragma unroll
  for (int j = 0; j < 4; ++j) {
    float bcol = bias ? bias[n0 + 16 * j + col] : 0.0f;
    float* Cp = Cbase + n0 + 16 * j;
#pragma unroll
    for (int r = 0; r < 8; ++r) {
      float v = acc[j][r] * alpha + bcol;
      if (relu) v = fmaxf(v, 0.0f);
      Cp[(long)r * ldc] = v;
    }
  }
}

// --- masked softmax over rows of length L (L <= 256), mask per batch --------
__global__ void __launch_bounds__(128)
masked_softmax_k(const float* __restrict__ in, float* __restrict__ out,
                 const mask_t* __restrict__ mask, int rowsPerBatch, int L,
                 float negval) {
  __shared__ float red[128];
  int tid = threadIdx.x;
  long row = blockIdx.x;
  long b = row / rowsPerBatch;
  const float* x = in + row * L;
  float* y = out + row * L;
  const mask_t* m = mask + b * L;

  float v0 = -INFINITY, v1 = -INFINITY;
  int i0 = tid, i1 = tid + 128;
  if (i0 < L) { v0 = x[i0]; if (m[i0]) v0 = negval; }
  if (i1 < L) { v1 = x[i1]; if (m[i1]) v1 = negval; }

  red[tid] = fmaxf(v0, v1);
  __syncthreads();
  for (int s = 64; s > 0; s >>= 1) {
    if (tid < s) red[tid] = fmaxf(red[tid], red[tid + s]);
    __syncthreads();
  }
  float mx = red[0];
  __syncthreads();

  float s0 = 0.f, s1 = 0.f;
  if (i0 < L) { v0 = __expf(v0 - mx); s0 = v0; }
  if (i1 < L) { v1 = __expf(v1 - mx); s1 = v1; }
  red[tid] = s0 + s1;
  __syncthreads();
  for (int s = 64; s > 0; s >>= 1) {
    if (tid < s) red[tid] += red[tid + s];
    __syncthreads();
  }
  float inv = 1.0f / red[0];
  if (i0 < L) y[i0] = v0 * inv;
  if (i1 < L) y[i1] = v1 * inv;
}

// --- logits[row] = H[row,:] . w + b2  (w2 is 512x1 GEMV) --------------------
__global__ void __launch_bounds__(128)
rowdot_k(const float* __restrict__ Hm, const float* __restrict__ w,
         const float* __restrict__ b2, float* __restrict__ out, int K) {
  __shared__ float red[128];
  long row = blockIdx.x;
  const float* h = Hm + row * K;
  float s = 0.f;
  for (int i = threadIdx.x; i < K; i += 128) s += h[i] * w[i];
  red[threadIdx.x] = s;
  __syncthreads();
  for (int st = 64; st > 0; st >>= 1) {
    if (threadIdx.x < st) red[threadIdx.x] += red[threadIdx.x + st];
    __syncthreads();
  }
  if (threadIdx.x == 0) out[row] = red[0] + b2[0];
}

// --- pooled[b,h] = sum_l att[b,l] * X[b,l,h]  -------------------------------
__global__ void __launch_bounds__(256)
weighted_pool_k(const float* __restrict__ X, const float* __restrict__ att,
                float* __restrict__ out, int L, int Hd) {
  int b = blockIdx.y;
  int h = blockIdx.x * 256 + threadIdx.x;
  const float* xb = X + (long)b * L * Hd + h;
  const float* ab = att + (long)b * L;
  float acc = 0.f;
  for (int l = 0; l < L; ++l) acc += ab[l] * xb[(long)l * Hd];
  out[(long)b * Hd + h] = acc;
}

__global__ void __launch_bounds__(256)
add2_k(const float* __restrict__ a, const float* __restrict__ b,
       float* __restrict__ c, int n) {
  int i = blockIdx.x * 256 + threadIdx.x;
  if (i < n) c[i] = a[i] + b[i];
}

// ---------------------------------------------------------------------------

extern "C" void kernel_launch(void* const* d_in, const int* in_sizes, int n_in,
                              void* d_out, int out_size, void* d_ws, size_t ws_size,
                              hipStream_t stream) {
  (void)in_sizes; (void)n_in; (void)out_size; (void)ws_size;

  const float*  i_batch = (const float*)d_in[0];   // [256,256,512]
  const float*  q_batch = (const float*)d_in[1];   // [256,64,512]
  const mask_t* i_mask  = (const mask_t*)d_in[2];  // [256,256] bool
  const mask_t* q_mask  = (const mask_t*)d_in[3];  // [256,64]  bool
  const float*  lf_w1   = (const float*)d_in[4];
  const float*  lf_b1   = (const float*)d_in[5];
  const float*  lf_w2   = (const float*)d_in[6];
  const float*  lf_b2   = (const float*)d_in[7];
  const float*  lf_wm   = (const float*)d_in[8];
  const float*  lf_bm   = (const float*)d_in[9];
  const float*  if_w1   = (const float*)d_in[10];
  const float*  if_b1   = (const float*)d_in[11];
  const float*  if_w2   = (const float*)d_in[12];
  const float*  if_b2   = (const float*)d_in[13];
  const float*  if_wm   = (const float*)d_in[14];
  const float*  if_bm   = (const float*)d_in[15];
  const float*  fh_w    = (const float*)d_in[16];
  const float*  fh_b    = (const float*)d_in[17];

  const int B = 256, IL = 256, QL = 64, H = 512, MID = 512, FO = 1024;

  // workspace layout (floats)
  float* ws      = (float*)d_ws;
  float* scores1 = ws;                                 // B*IL*QL
  float* ifeat   = scores1 + (long)B * IL * QL;        // B*IL*H
  float* Hbuf    = ifeat   + (long)B * IL * H;         // (B*IL)*MID (reused)
  float* logits  = Hbuf    + (long)B * IL * MID;       // B*IL (reused)
  float* scores2 = logits  + (long)B * IL;             // B*QL*IL
  float* lfeat   = scores2 + (long)B * QL * IL;        // B*QL*H
  float* att2    = lfeat   + (long)B * QL * H;         // B*QL
  float* pooled1 = att2    + (long)B * QL;             // B*H
  float* pooled2 = pooled1 + (long)B * H;              // B*H
  float* F1      = pooled2 + (long)B * H;              // B*FO
  float* F2      = F1      + (long)B * FO;             // B*FO
  float* Fsum    = F2      + (long)B * FO;             // B*FO

  float* outp     = (float*)d_out;                     // [B,512]
  float* i_weight = outp + (long)B * 512;              // [B,256] (att1)

  const float scale = 0.044194173824159216f;           // 1/sqrt(512)
  dim3 blk(128);

  // ---- branch 1: i_feat = attn(i_batch, q_batch, q_batch, q_mask) ----------
  // scores1 = (i_batch @ q_batch^T) * scale        [B,256,64]
  wmma_gemm<1><<<dim3(QL / 64, IL / 64, B), blk, 0, stream>>>(
      i_batch, q_batch, nullptr, scores1, H,
      (long)IL * H, (long)QL * H, (long)IL * QL, H, H, QL, scale, 0);
  masked_softmax_k<<<B * IL, blk, 0, stream>>>(scores1, scores1, q_mask, IL, QL,
                                               -65504.0f);
  // ifeat = P1 @ q_batch                           [B,256,512]
  wmma_gemm<0><<<dim3(H / 64, IL / 64, B), blk, 0, stream>>>(
      scores1, q_batch, nullptr, ifeat, QL,
      (long)IL * QL, (long)QL * H, (long)IL * H, QL, H, H, 1.0f, 0);
  // Hbuf = relu(ifeat @ lf_w1 + lf_b1)             [65536,512]
  wmma_gemm<0><<<dim3(MID / 64, (B * IL) / 64, 1), blk, 0, stream>>>(
      ifeat, lf_w1, lf_b1, Hbuf, H, 0, 0, 0, H, MID, MID, 1.0f, 1);
  rowdot_k<<<B * IL, blk, 0, stream>>>(Hbuf, lf_w2, lf_b2, logits, MID);
  // att1 -> i_weight slice of d_out                [B,256]
  masked_softmax_k<<<B, blk, 0, stream>>>(logits, i_weight, i_mask, 1, IL, -1e9f);
  weighted_pool_k<<<dim3(H / 256, B), 256, 0, stream>>>(ifeat, i_weight, pooled1,
                                                        IL, H);
  // F1 = pooled1 @ lf_wm + lf_bm                   [B,1024]
  wmma_gemm<0><<<dim3(FO / 64, B / 64, 1), blk, 0, stream>>>(
      pooled1, lf_wm, lf_bm, F1, H, 0, 0, 0, H, FO, FO, 1.0f, 0);

  // ---- branch 2: l_feat = attn(q_batch, i_batch, i_batch, i_mask) ----------
  // scores2 = (q_batch @ i_batch^T) * scale        [B,64,256]
  wmma_gemm<1><<<dim3(IL / 64, QL / 64, B), blk, 0, stream>>>(
      q_batch, i_batch, nullptr, scores2, H,
      (long)QL * H, (long)IL * H, (long)QL * IL, H, H, IL, scale, 0);
  masked_softmax_k<<<B * QL, blk, 0, stream>>>(scores2, scores2, i_mask, QL, IL,
                                               -65504.0f);
  // lfeat = P2 @ i_batch                           [B,64,512]
  wmma_gemm<0><<<dim3(H / 64, QL / 64, B), blk, 0, stream>>>(
      scores2, i_batch, nullptr, lfeat, IL,
      (long)QL * IL, (long)IL * H, (long)QL * H, IL, H, H, 1.0f, 0);
  // Hbuf = relu(lfeat @ if_w1 + if_b1)             [16384,512]
  wmma_gemm<0><<<dim3(MID / 64, (B * QL) / 64, 1), blk, 0, stream>>>(
      lfeat, if_w1, if_b1, Hbuf, H, 0, 0, 0, H, MID, MID, 1.0f, 1);
  rowdot_k<<<B * QL, blk, 0, stream>>>(Hbuf, if_w2, if_b2, logits, MID);
  masked_softmax_k<<<B, blk, 0, stream>>>(logits, att2, q_mask, 1, QL, -1e9f);
  weighted_pool_k<<<dim3(H / 256, B), 256, 0, stream>>>(lfeat, att2, pooled2,
                                                        QL, H);
  // F2 = pooled2 @ if_wm + if_bm                   [B,1024]
  wmma_gemm<0><<<dim3(FO / 64, B / 64, 1), blk, 0, stream>>>(
      pooled2, if_wm, if_bm, F2, H, 0, 0, 0, H, FO, FO, 1.0f, 0);

  // ---- head: out = (F1 + F2) @ fh_w + fh_b       [B,512] ------------------
  add2_k<<<(B * FO + 255) / 256, 256, 0, stream>>>(F1, F2, Fsum, B * FO);
  wmma_gemm<0><<<dim3(512 / 64, B / 64, 1), blk, 0, stream>>>(
      Fsum, fh_w, fh_b, outp, FO, 0, 0, 0, FO, 512, 512, 1.0f, 0);
}
